// SparseMultiHeadAttention_79259326481042
// MI455X (gfx1250) — matrix operand order repaired and checked by reading, not compiled, not run on hardware
//
#include <hip/hip_runtime.h>
#include <hip/hip_bf16.h>
#include <stdint.h>

// ---------------------------------------------------------------------------
// Problem constants (match reference): B=2, L=2048, E=1024, H=8, D=128
// ---------------------------------------------------------------------------
constexpr int BB = 2;
constexpr int LL = 2048;
constexpr int EE = 1024;
constexpr int HH = 8;
constexpr int DD = 128;
constexpr int MTOT = BB * LL;   // 4096 rows for all projections

// ---------------------------------------------------------------------------
// WMMA / TDM types (CDNA5 gfx1250, wave32)
// ---------------------------------------------------------------------------
typedef __attribute__((ext_vector_type(16))) __bf16 v16bf;
typedef __attribute__((ext_vector_type(8)))  __bf16 v8bf;
typedef __attribute__((ext_vector_type(8)))  float  v8f;
typedef __attribute__((ext_vector_type(4)))  unsigned int u32x4;
typedef __attribute__((ext_vector_type(8)))  int          i32x8;
typedef __attribute__((ext_vector_type(4)))  int          i32x4;

__device__ inline v8f wmma_bf16(v16bf a, v16bf b, v8f c) {
  return __builtin_amdgcn_wmma_f32_16x16x32_bf16(
      false, a, false, b, (short)0, c, false, false);
}

// float -> bf16 round-to-nearest-even
__device__ inline __bf16 f2bf(float x) {
  union { float f; unsigned u; } cv; cv.f = x;
  unsigned u = cv.u;
  u += 0x7FFFu + ((u >> 16) & 1u);
  union { unsigned short s; __bf16 h; } r;
  r.s = (unsigned short)(u >> 16);
  return r.h;
}

// Generic shared pointer -> LDS byte offset (flat LDS aperture keeps the
// LDS offset in addr[31:0]).
__device__ inline unsigned int lds_off(const void* p) {
  return (unsigned int)(uintptr_t)p;
}

// ---------------------------------------------------------------------------
// Tensor Data Mover: async DMA of a 2D bf16 tile (tile_d1 rows x tile_d0 cols,
// row stride `stride` elements) from global memory into LDS at `ldsb`.
// D# built per CDNA5 ISA ch.8:
//  group0: count=1 | lds_addr | global_addr[56:0] | type=2
//  group1: data_size=2B, pad_enable, pad_interval=16 DWORDs, pad_amount=4
//          DWORDs  -> LDS row stride = 64B + 16B = 80B = 40 bf16 (our LDT),
//          tensor dims oversized (tiles never go OOB), dim0_stride = K.
// Tracked by TENSORcnt; EXEC is ignored so only wave 0 issues it.
// ---------------------------------------------------------------------------
__device__ inline void tdm_load_tile(unsigned int ldsb, const __bf16* g,
                                     int tile_d0, int tile_d1,
                                     long long stride) {
  unsigned long long ga = (unsigned long long)(uintptr_t)g;
  u32x4 g0;
  g0[0] = 1u;                                   // count=1 (valid user D#)
  g0[1] = ldsb;                                 // lds_addr
  g0[2] = (unsigned int)ga;                     // global_addr[31:0]
  g0[3] = (unsigned int)((ga >> 32) & 0x1FFFFFFu) | (2u << 30);  // [56:32]|type=2
  i32x8 g1;
  g1[0] = (1 << 16) | (1 << 20) | (3 << 22) | (3 << 25);
  //        data_size=2B  pad_en   interval=16dw  amount=4dw ; wg_mask=0
  const int td0 = 1 << 20, td1 = 1 << 20;       // oversized tensor dims
  g1[1] = (td0 & 0xFFFF) << 16;                 // abar_addr=0 | tensor_dim0 lo
  g1[2] = ((td0 >> 16) & 0xFFFF) | ((td1 & 0xFFFF) << 16);
  g1[3] = ((td1 >> 16) & 0xFFFF) | ((tile_d0 & 0xFFFF) << 16);   // tile_dim0
  g1[4] = (tile_d1 & 0xFFFF);                   // tile_dim1, tile_dim2=0
  g1[5] = (int)(unsigned int)(stride & 0xFFFFFFFFll);            // dim0_stride lo
  g1[6] = (int)(unsigned int)((stride >> 32) & 0xFFFFll);        // hi16, dim1_stride=0
  g1[7] = 0;
  i32x4 z4 = {0, 0, 0, 0};
#if defined(__clang_major__) && (__clang_major__ >= 23)
  i32x8 z8 = {0, 0, 0, 0, 0, 0, 0, 0};
  __builtin_amdgcn_tensor_load_to_lds(g0, g1, z4, z4, z8, 0);
#else
  __builtin_amdgcn_tensor_load_to_lds(g0, g1, z4, z4, 0);
#endif
}

// A fragment: 16x32 bf16, rows striped per ISA table:
//   lanes 0-15  : row = lane,    K elems {0..7, 16..23}
//   lanes 16-31 : row = lane-16, K elems {8..15, 24..31}
__device__ inline v16bf load_frag_a(const __bf16* p, int ld) {
  const int l = threadIdx.x & 31;
  const __bf16* rp = p + (size_t)(l & 15) * ld + ((l >> 4) << 3);
  v8bf lo = *(const v8bf*)rp;
  v8bf hi = *(const v8bf*)(rp + 16);
  v16bf f;
#pragma unroll
  for (int i = 0; i < 8; ++i) { f[i] = lo[i]; f[i + 8] = hi[i]; }
  return f;
}

// B fragment: 32x16 bf16 (K x N): lanes 0-15 col=lane K={0..7,16..23},
// lanes 16-31 col=lane-16 K={8..15,24..31}. p at (n0,k0) of row-major [N][ld] = B^T.
__device__ inline v16bf load_frag_b(const __bf16* p, int ld) {
  const int l = threadIdx.x & 31;
  const __bf16* rp = p + (size_t)(l & 15) * ld + ((l >> 4) << 3);
  v8bf lo = *(const v8bf*)rp;
  v8bf hi = *(const v8bf*)(rp + 16);
  v16bf f;
#pragma unroll
  for (int i = 0; i < 8; ++i) { f[i] = lo[i]; f[i + 8] = hi[i]; }
  return f;
}

__device__ inline float hmax16(float x) {
#pragma unroll
  for (int m = 1; m < 16; m <<= 1) x = fmaxf(x, __shfl_xor(x, m, 32));
  return x;
}
__device__ inline float hsum16(float x) {
#pragma unroll
  for (int m = 1; m < 16; m <<= 1) x += __shfl_xor(x, m, 32);
  return x;
}

// anchors: unique(int(r * (L-1))) -- L constexpr, folds to integer compares
__device__ inline bool is_anchor(int j) {
  const float rr[11] = {0.0f, 0.1f, 0.2f, 0.3f, 0.382f, 0.5f,
                        0.618f, 0.7f, 0.8f, 0.9f, 1.0f};
#pragma unroll
  for (int t = 0; t < 11; ++t)
    if (j == (int)(rr[t] * (float)(LL - 1))) return true;
  return false;
}

// ---------------------------------------------------------------------------
// fp32 -> bf16 elementwise conversion (activations + weights, done once so
// the GEMM tiles are TDM-DMA-able without conversion).
// ---------------------------------------------------------------------------
__global__ __launch_bounds__(256) void cvt_f32_bf16(const float* __restrict__ s,
                                                    __bf16* __restrict__ d,
                                                    int n) {
  int i = blockIdx.x * 1024 + threadIdx.x;
#pragma unroll
  for (int k = 0; k < 4; ++k, i += 256)
    if (i < n) d[i] = f2bf(s[i]);
}

// ---------------------------------------------------------------------------
// TDM-fed projection GEMM: Y[m,n] = sum_k X[m,k]*W[n,k] + bias[n]
// X bf16 [M,K], W bf16 [N,K] (torch Linear layout), bias fp32.
// 128x128 tile / 256-thread WG (8 waves, each 32Mx64N).
// Double-buffered: wave0 issues tensor_load_to_lds for tile it+1, waits
// TENSORcnt<=2 (=> tile it landed), barrier releases the other 7 waves.
// TDM padding gives LDS row stride 40 bf16 (80B) matching the frag loaders.
// MODE 0: out bf16 [B,H,L,D] (Q,K) | MODE 1: bf16 [B,H,D,L] (V^T)
// MODE 2: out fp32 [M,N] (final projection -> d_out)
// ---------------------------------------------------------------------------
template <int MODE>
__global__ __launch_bounds__(256) void proj_gemm(const __bf16* __restrict__ X,
                                                 const __bf16* __restrict__ W,
                                                 const float* __restrict__ bias,
                                                 void* __restrict__ out) {
  constexpr int K = EE;
  constexpr int LDT = 40;       // 32 cols + 4-DWORD TDM pad
  constexpr int NIT = K / 32;   // 32 K-steps
  __shared__ __align__(16) __bf16 As[2][128 * LDT];
  __shared__ __align__(16) __bf16 Bs[2][128 * LDT];

  const int m0 = blockIdx.x * 128;
  const int n0 = blockIdx.y * 128;
  const int t = threadIdx.x;
  const int wv = t >> 5;
  const int wm = wv & 3;   // 4 waves along M
  const int wn = wv >> 2;  // 2 waves along N
  const bool issuer = (t < 32);  // wave 0 drives the TDM

  v8f acc[2][4] = {};

  if (issuer) {
    tdm_load_tile(lds_off(&As[0][0]), X + (size_t)m0 * K, 32, 128, K);
    tdm_load_tile(lds_off(&Bs[0][0]), W + (size_t)n0 * K, 32, 128, K);
  }

  for (int it = 0; it < NIT; ++it) {
    if (issuer) {
      if (it + 1 < NIT) {
        const int k1 = (it + 1) * 32;
        tdm_load_tile(lds_off(&As[(it + 1) & 1][0]),
                      X + (size_t)m0 * K + k1, 32, 128, K);
        tdm_load_tile(lds_off(&Bs[(it + 1) & 1][0]),
                      W + (size_t)n0 * K + k1, 32, 128, K);
        __builtin_amdgcn_s_wait_tensorcnt(2);   // tile `it` complete
      } else {
        __builtin_amdgcn_s_wait_tensorcnt(0);
      }
    }
    __syncthreads();   // LDS tile `it` visible to all 8 waves

    const __bf16* Ab = &As[it & 1][0];
    const __bf16* Bb = &Bs[it & 1][0];
    v16bf a0 = load_frag_a(Ab + (wm * 32 + 0) * LDT, LDT);
    v16bf a1 = load_frag_a(Ab + (wm * 32 + 16) * LDT, LDT);
#pragma unroll
    for (int nt = 0; nt < 4; ++nt) {
      v16bf bf = load_frag_b(Bb + (wn * 64 + nt * 16) * LDT, LDT);
      acc[0][nt] = wmma_bf16(a0, bf, acc[0][nt]);
      acc[1][nt] = wmma_bf16(a1, bf, acc[1][nt]);
    }
    __syncthreads();   // buffer reusable by the TDM next-next iteration
  }

  // epilogue: C-frag layout (lanes 0-15 = cols, VGPR r = row; lanes>=16: +8)
  const int lane = t & 31;
  const int col = lane & 15;
  const int hal = lane >> 4;
#pragma unroll
  for (int mt = 0; mt < 2; ++mt) {
#pragma unroll
    for (int nt = 0; nt < 4; ++nt) {
#pragma unroll
      for (int r = 0; r < 8; ++r) {
        int m = m0 + wm * 32 + mt * 16 + r + hal * 8;
        int n = n0 + wn * 64 + nt * 16 + col;
        float v = acc[mt][nt][r] + bias[n];
        if (MODE == 2) {
          ((float*)out)[(size_t)m * EE + n] = v;
        } else {
          int b = m >> 11;        // / L
          int l = m & (LL - 1);
          int h = n >> 7;         // / D
          int d = n & (DD - 1);
          __bf16 bvv = f2bf(v);
          if (MODE == 0)
            ((__bf16*)out)[(((size_t)b * HH + h) * LL + l) * DD + d] = bvv;
          else
            ((__bf16*)out)[(((size_t)b * HH + h) * DD + d) * LL + l] = bvv;
        }
      }
    }
  }
}

// ---------------------------------------------------------------------------
// Block-sparse flash attention (1 wave / 16 query rows, 32-wide KV blocks).
// Head -> pattern: h0 |i-j|<=8 ; h1,h2 |i-j|<=16 ; h3..5 |i-j|<=32 or j%8==0 ;
// h6,7 j%16==0 or anchor(j) ; all AND causal j<=i.
// Writes ctx directly in bf16 [M, E] to feed the TDM-fed output projection.
// ---------------------------------------------------------------------------
__global__ __launch_bounds__(32) void sparse_attn(const __bf16* __restrict__ q,
                                                  const __bf16* __restrict__ k,
                                                  const __bf16* __restrict__ vt,
                                                  __bf16* __restrict__ ctx) {
  constexpr int NT = LL / 16;
  const int tile = blockIdx.x;
  const int i0 = (tile % NT) * 16;
  const int h = (tile / NT) % HH;
  const int b = tile / (NT * HH);

  const int lane = threadIdx.x & 31;
  const int col = lane & 15;
  const int hal = lane >> 4;

  int win, cm;  // cm: 0 window only, 1 window+stride8, 2 stride16+anchors
  if (h == 0)      { win = 8;  cm = 0; }
  else if (h <= 2) { win = 16; cm = 0; }
  else if (h <= 5) { win = 32; cm = 1; }
  else             { win = -1; cm = 2; }

  const __bf16* qb = q + ((size_t)b * HH + h) * LL * DD;
  const __bf16* kb = k + ((size_t)b * HH + h) * LL * DD;
  const __bf16* vb = vt + ((size_t)b * HH + h) * DD * LL;

  v16bf qf[4];
#pragma unroll
  for (int kk = 0; kk < 4; ++kk)
    qf[kk] = load_frag_a(qb + (size_t)i0 * DD + kk * 32, DD);

  v8f o[8] = {};
  float rowM[8], rowS[8];
#pragma unroll
  for (int r = 0; r < 8; ++r) { rowM[r] = -1e30f; rowS[r] = 0.f; }

  const int jbEnd = (i0 + 15) >> 5;
  int jbStart = 0;
  if (cm == 0) {
    int lo = i0 - win;
    jbStart = lo > 0 ? (lo >> 5) : 0;
  }

  __shared__ __align__(16) __bf16 pb[16 * 32];
  const float scale = 0.088388347648318447f;  // 1/sqrt(128)

  for (int jb = jbStart; jb <= jbEnd; ++jb) {
    const int j0 = jb << 5;

    if (jb < jbEnd) {  // global_prefetch_b8 next K/V block
      __builtin_prefetch(kb + (size_t)(j0 + 32 + lane) * DD, 0, 0);
      __builtin_prefetch(vb + (size_t)(lane * 4) * LL + j0 + 32, 0, 0);
    }

    v8f s[2] = {};
#pragma unroll
    for (int nt = 0; nt < 2; ++nt) {
#pragma unroll
      for (int kk = 0; kk < 4; ++kk) {
        v16bf bf = load_frag_b(kb + (size_t)(j0 + nt * 16) * DD + kk * 32, DD);
        s[nt] = wmma_bf16(qf[kk], bf, s[nt]);
      }
    }

#pragma unroll
    for (int nt = 0; nt < 2; ++nt) {
      int j = j0 + nt * 16 + col;
      bool colOK =
          (cm == 1) ? ((j & 7) == 0)
                    : ((cm == 2) ? (((j & 15) == 0) || is_anchor(j)) : false);
#pragma unroll
      for (int r = 0; r < 8; ++r) {
        int i = i0 + r + hal * 8;
        bool ok = (j <= i) &&
                  ((cm == 2) ? colOK : (((i - j) <= win) || colOK));
        float sv = s[nt][r] * scale;
        s[nt][r] = ok ? sv : -1e30f;
      }
    }

    float pf[2][8];
#pragma unroll
    for (int r = 0; r < 8; ++r) {
      float mx = hmax16(fmaxf(s[0][r], s[1][r]));
      float newM = fmaxf(rowM[r], mx);
      float corr = __expf(rowM[r] - newM);
      float ls = 0.f;
#pragma unroll
      for (int nt = 0; nt < 2; ++nt) {
        float sv = s[nt][r];
        float p = (sv > -5e29f) ? __expf(sv - newM) : 0.f;
        pf[nt][r] = p;
        ls += p;
      }
      ls = hsum16(ls);
      rowS[r] = rowS[r] * corr + ls;
      rowM[r] = newM;
#pragma unroll
      for (int dt = 0; dt < 8; ++dt) o[dt][r] *= corr;
    }

    // C-layout -> A-layout round trip via LDS (single wave, DS in-order)
#pragma unroll
    for (int nt = 0; nt < 2; ++nt)
#pragma unroll
      for (int r = 0; r < 8; ++r)
        pb[(r + hal * 8) * 32 + nt * 16 + col] = f2bf(pf[nt][r]);

    v16bf pfrag = load_frag_a(pb, 32);
#pragma unroll
    for (int dt = 0; dt < 8; ++dt) {
      v16bf vfr = load_frag_b(vb + (size_t)(dt * 16) * LL + j0, LL);
      o[dt] = wmma_bf16(pfrag, vfr, o[dt]);
    }
  }

#pragma unroll
  for (int r = 0; r < 8; ++r) {
    float inv = rowS[r] > 0.f ? 1.f / rowS[r] : 0.f;
    int i = i0 + r + hal * 8;
#pragma unroll
    for (int dt = 0; dt < 8; ++dt)
      ctx[((size_t)(b * LL + i)) * EE + h * DD + dt * 16 + col] =
          f2bf(o[dt][r] * inv);
  }
}

// ---------------------------------------------------------------------------
// Launch: f32->bf16 converts -> 3 TDM-fed WMMA projections -> block-sparse
// flash attention (WMMA) -> TDM-fed WMMA output projection.
// Inputs: 0 query, 1 key, 2 value, 3 Wq, 4 bq, 5 Wk, 6 bk, 7 Wv, 8 bv,
//         9 Wo, 10 bo (fp32). Output fp32 [B,L,E].
// Workspace (bf16): Xq|Xk|Xv (3x8MB) Wq..Wo (4x2MB) Qb|Kb|V^T (3x8MB)
//                   ctx (8MB) = 64MB.
// ---------------------------------------------------------------------------
extern "C" void kernel_launch(void* const* d_in, const int* in_sizes, int n_in,
                              void* d_out, int out_size, void* d_ws,
                              size_t ws_size, hipStream_t stream) {
  (void)in_sizes; (void)n_in; (void)out_size; (void)ws_size;

  constexpr size_t NX = (size_t)MTOT * EE;  // 4M elems
  constexpr size_t NW = (size_t)EE * EE;    // 1M elems

  __bf16* base = (__bf16*)d_ws;
  __bf16* xq = base;                 // activations
  __bf16* xk = xq + NX;
  __bf16* xv = xk + NX;
  __bf16* wq = xv + NX;              // weights
  __bf16* wk = wq + NW;
  __bf16* wvp = wk + NW;
  __bf16* wo = wvp + NW;
  __bf16* qb = wo + NW;              // projected Q [B,H,L,D]
  __bf16* kb = qb + NX;              // projected K [B,H,L,D]
  __bf16* vt = kb + NX;              // projected V^T [B,H,D,L]
  __bf16* ctx = vt + NX;             // attention output [M,E]

  const int gx = (int)((NX + 1023) / 1024);
  const int gw = (int)((NW + 1023) / 1024);
  cvt_f32_bf16<<<gx, 256, 0, stream>>>((const float*)d_in[0], xq, (int)NX);
  cvt_f32_bf16<<<gx, 256, 0, stream>>>((const float*)d_in[1], xk, (int)NX);
  cvt_f32_bf16<<<gx, 256, 0, stream>>>((const float*)d_in[2], xv, (int)NX);
  cvt_f32_bf16<<<gw, 256, 0, stream>>>((const float*)d_in[3], wq, (int)NW);
  cvt_f32_bf16<<<gw, 256, 0, stream>>>((const float*)d_in[5], wk, (int)NW);
  cvt_f32_bf16<<<gw, 256, 0, stream>>>((const float*)d_in[7], wvp, (int)NW);
  cvt_f32_bf16<<<gw, 256, 0, stream>>>((const float*)d_in[9], wo, (int)NW);

  dim3 gg(MTOT / 128, EE / 128);  // 32 x 8 workgroups
  proj_gemm<0><<<gg, 256, 0, stream>>>(xq, wq, (const float*)d_in[4], qb);
  proj_gemm<0><<<gg, 256, 0, stream>>>(xk, wk, (const float*)d_in[6], kb);
  proj_gemm<1><<<gg, 256, 0, stream>>>(xv, wvp, (const float*)d_in[8], vt);

  sparse_attn<<<BB * HH * (LL / 16), 32, 0, stream>>>(qb, kb, vt, ctx);

  proj_gemm<2><<<gg, 256, 0, stream>>>(ctx, wo, (const float*)d_in[10], d_out);
}
